// encoder_unit_80109730005719
// MI455X (gfx1250) — matrix operand order repaired and checked by reading, not compiled, not run on hardware
//
#include <hip/hip_runtime.h>

// ---------------------------------------------------------------------------
// Problem constants (B=2, S=2048, D=1024, H=16, DH=64)
// ---------------------------------------------------------------------------
#define BATCH 2
#define SEQ   2048
#define DMODEL 1024
#define NHEAD 16
#define DHEAD 64
#define MROWS (BATCH*SEQ)          // 4096 token rows
#define DFF   (4*DMODEL)           // 4096

typedef __attribute__((ext_vector_type(16))) __bf16 v16bf;
typedef __attribute__((ext_vector_type(4)))  __bf16 v4bf;
typedef __attribute__((ext_vector_type(8)))  float  v8f;
typedef __attribute__((ext_vector_type(4)))  unsigned int v4u;
typedef __attribute__((ext_vector_type(8)))  int v8i;
typedef __attribute__((ext_vector_type(4)))  int v4i;

union FragBF { v16bf v; uint4 u[2]; };

__device__ __forceinline__ v8f wmma_bf16(v16bf a, v16bf b, v8f c) {
  // (neg_a, A, neg_b, B, c_mod, C, reuse_a, reuse_b)
  return __builtin_amdgcn_wmma_f32_16x16x32_bf16(false, a, false, b, (short)0, c,
                                                 false, false);
}
__device__ __forceinline__ v8f v8f_zero() {
  v8f z = {0.f, 0.f, 0.f, 0.f, 0.f, 0.f, 0.f, 0.f};
  return z;
}

// 16-lane xor-tree reductions as DPP (pure VALU; avoids ds_bpermute storms).
// quad_perm xor1 = 0xB1, quad_perm xor2 = 0x4E, row_half_mirror = 0x141,
// row_mirror = 0x140. Network: after the 4 steps all 16 lanes hold the result.
#define DPPF(v, ctrl)                                                          \
  __int_as_float(__builtin_amdgcn_update_dpp(0, __float_as_int(v), ctrl, 0xf,  \
                                             0xf, true))
__device__ __forceinline__ float red16_max(float v) {
  v = fmaxf(v, DPPF(v, 0xB1));
  v = fmaxf(v, DPPF(v, 0x4E));
  v = fmaxf(v, DPPF(v, 0x141));
  v = fmaxf(v, DPPF(v, 0x140));
  return v;
}
__device__ __forceinline__ float red16_sum(float v) {
  v += DPPF(v, 0xB1);
  v += DPPF(v, 0x4E);
  v += DPPF(v, 0x141);
  v += DPPF(v, 0x140);
  return v;
}

// ---------------------------------------------------------------------------
// Tensor Data Mover: 2D bf16 tile load Global->LDS with LDS row padding.
// tile_dim0 = 32 elems (64B rows), pad 16B after each 64B -> 80B LDS pitch,
// matching the conflict-free 40-element LDS rows used by the WMMA frag reads.
// ---------------------------------------------------------------------------
#if __has_builtin(__builtin_amdgcn_tensor_load_to_lds) && \
    __has_builtin(__builtin_amdgcn_s_wait_tensorcnt)
#define HAVE_TDM 1
#else
#define HAVE_TDM 0
#endif

#if HAVE_TDM
__device__ __forceinline__ void tdm_load_tile32(const void* gsrc,
                                                const void* lds_dst,
                                                unsigned tensor_rows,
                                                unsigned row_elems,
                                                unsigned tile_rows) {
  unsigned long long ga = (unsigned long long)gsrc;
  unsigned lds_off = (unsigned)(unsigned long long)lds_dst;  // low 32 = LDS byte offset
  v4u g0;
  g0[0] = 1u;                                   // count=1 (valid), user mode
  g0[1] = lds_off;                              // lds_addr
  g0[2] = (unsigned)(ga & 0xffffffffu);         // global_addr[31:0]
  g0[3] = (unsigned)((ga >> 32) & 0x01ffffffu) | (2u << 30);  // addr[56:32], type=2
  v8i g1;
  g1[0] = (int)((1u << 16)        // data_size = 1 -> 2 bytes
              | (1u << 20)        // pad_enable
              | (3u << 22)        // pad_interval: 16 DWORDs (64B)
              | (3u << 25));      // pad_amount:   4 DWORDs (16B)
  g1[1] = (int)((row_elems & 0xffffu) << 16);                    // tensor_dim0 lo
  g1[2] = (int)(((row_elems >> 16) & 0xffffu) |
                ((tensor_rows & 0xffffu) << 16));                // dim0 hi | dim1 lo
  g1[3] = (int)(((tensor_rows >> 16) & 0xffffu) | (32u << 16));  // dim1 hi | tile_dim0
  g1[4] = (int)(tile_rows & 0xffffu);                            // tile_dim1, tile_dim2=0
  g1[5] = (int)row_elems;                                        // dim0_stride[31:0]
  g1[6] = 0;                                                     // stride hi | dim1_stride lo
  g1[7] = 0;
  v4i g2; g2[0] = 1; g2[1] = 0; g2[2] = 0; g2[3] = 0;            // tensor_dim2=1 (unused)
  v4i g3; g3[0] = 0; g3[1] = 0; g3[2] = 0; g3[3] = 0;
#if __clang_major__ >= 23
  v8i gz; gz[0]=0; gz[1]=0; gz[2]=0; gz[3]=0; gz[4]=0; gz[5]=0; gz[6]=0; gz[7]=0;
  __builtin_amdgcn_tensor_load_to_lds(g0, g1, g2, g3, gz, 0);
#else
  __builtin_amdgcn_tensor_load_to_lds(g0, g1, g2, g3, 0);
#endif
}
#endif

// ---------------------------------------------------------------------------
// f32 -> bf16 conversion (vectorized, n must be multiple of 4)
// ---------------------------------------------------------------------------
__global__ void f32_to_bf16_k(const float* __restrict__ in,
                              __bf16* __restrict__ out, long n4) {
  long i = (long)blockIdx.x * blockDim.x + threadIdx.x;
  if (i >= n4) return;
  float4 f = ((const float4*)in)[i];
  v4bf o;
  o[0] = (__bf16)f.x; o[1] = (__bf16)f.y; o[2] = (__bf16)f.z; o[3] = (__bf16)f.w;
  ((v4bf*)out)[i] = o;
}

__global__ void copy_f32_k(const float* __restrict__ in, float* __restrict__ out,
                           long n) {
  long i = (long)blockIdx.x * blockDim.x + threadIdx.x;
  if (i < n) out[i] = in[i];
}

// ---------------------------------------------------------------------------
// GEMM:  C[M,N] = A_bf16[M,K] * W_bf16[N,K]^T + bias  (+ residual) (+ relu)
// Block tile 128x256, 8 waves (2 along M x 4 along N), wave tile 64x64,
// K-step 32. LDS rows padded to 40 bf16 for conflict-free b128 frag reads.
// TDM path: wave 0 drives double-buffered tensor_load_to_lds DMA (TENSORcnt),
// overlapping the next tile's DMA with the current tile's 16 WMMAs per wave.
// ---------------------------------------------------------------------------
__device__ __forceinline__ void gemm_frag_mma(const __bf16* sAp,
                                              const __bf16* sBp, int waveM,
                                              int waveN, int l15, int lhalf,
                                              v8f acc[4][4]) {
  FragBF af[4], bfr[4];
#pragma unroll
  for (int sm = 0; sm < 4; sm++) {
    const __bf16* p = sAp + (waveM + sm * 16 + l15) * 40 + lhalf * 8;
    af[sm].u[0] = *(const uint4*)p;         // K = base .. base+7
    af[sm].u[1] = *(const uint4*)(p + 16);  // K = base+16 .. base+23
  }
#pragma unroll
  for (int sn = 0; sn < 4; sn++) {
    const __bf16* p = sBp + (waveN + sn * 16 + l15) * 40 + lhalf * 8;
    bfr[sn].u[0] = *(const uint4*)p;        // K = base .. base+7
    bfr[sn].u[1] = *(const uint4*)(p + 16);  // K = base+16 .. base+23
  }
#pragma unroll
  for (int sm = 0; sm < 4; sm++)
#pragma unroll
    for (int sn = 0; sn < 4; sn++)
      acc[sm][sn] = wmma_bf16(af[sm].v, bfr[sn].v, acc[sm][sn]);
}

template <bool RELU, bool RES, bool OUTBF>
__global__ __launch_bounds__(256) void gemm_bf16_k(
    const __bf16* __restrict__ A, const __bf16* __restrict__ W,
    const float* __restrict__ bias, const float* __restrict__ res,
    float* __restrict__ Cf, __bf16* __restrict__ Cb, int M, int N, int K) {
  __shared__ __bf16 sA[2][128 * 40];
  __shared__ __bf16 sB[2][256 * 40];
  const int tid = threadIdx.x;
  const int wave = tid >> 5;
  const int lane = tid & 31;
  const int l15 = lane & 15;
  const int lhalf = lane >> 4;
  const int bm = blockIdx.x * 128;
  const int bn = blockIdx.y * 256;
  const int waveM = (wave & 1) * 64;
  const int waveN = (wave >> 1) * 64;

  v8f acc[4][4];
#pragma unroll
  for (int i = 0; i < 4; i++)
#pragma unroll
    for (int j = 0; j < 4; j++) acc[i][j] = v8f_zero();

#if HAVE_TDM
  const int nk = K / 32;
  if (wave == 0) {
    tdm_load_tile32(A + (size_t)bm * K, &sA[0][0], (unsigned)M, (unsigned)K, 128);
    tdm_load_tile32(W + (size_t)bn * K, &sB[0][0], (unsigned)N, (unsigned)K, 256);
  }
  for (int i = 0; i < nk; i++) {
    const int cur = i & 1, nxt = cur ^ 1;
    __syncthreads();  // buffer `nxt` free: everyone done with iteration i-1
    if (wave == 0) {
      if (i + 1 < nk) {
        tdm_load_tile32(A + (size_t)bm * K + (i + 1) * 32, &sA[nxt][0],
                        (unsigned)M, (unsigned)K, 128);
        tdm_load_tile32(W + (size_t)bn * K + (i + 1) * 32, &sB[nxt][0],
                        (unsigned)N, (unsigned)K, 256);
        __builtin_amdgcn_s_wait_tensorcnt((short)2);  // current tile's DMA done
      } else {
        __builtin_amdgcn_s_wait_tensorcnt((short)0);
      }
    }
    __syncthreads();  // current tile visible to all waves
    gemm_frag_mma(&sA[cur][0], &sB[cur][0], waveM, waveN, l15, lhalf, acc);
  }
#else
  // Fallback: manual staging, single buffer.
  const int rowS = tid >> 1;  // A staging row 0..127
  const int segS = tid & 1;   // which 32B half of the 64B row
  for (int k0 = 0; k0 < K; k0 += 32) {
    __syncthreads();
    {
      const uint4* ga =
          (const uint4*)(A + (size_t)(bm + rowS) * K + k0 + segS * 16);
      uint4 a0 = ga[0], a1 = ga[1];
      uint4* la = (uint4*)(&sA[0][0] + rowS * 40 + segS * 16);
      la[0] = a0; la[1] = a1;
      const uint4* gw = (const uint4*)(W + (size_t)(bn + tid) * K + k0);
      uint4 w0 = gw[0], w1 = gw[1], w2 = gw[2], w3 = gw[3];
      uint4* lw = (uint4*)(&sB[0][0] + tid * 40);
      lw[0] = w0; lw[1] = w1; lw[2] = w2; lw[3] = w3;
    }
    if (k0 + 32 < K) {
      __builtin_prefetch(A + (size_t)(bm + rowS) * K + k0 + 32, 0, 1);
      __builtin_prefetch(W + (size_t)(bn + tid) * K + k0 + 32, 0, 1);
    }
    __syncthreads();
    gemm_frag_mma(&sA[0][0], &sB[0][0], waveM, waveN, l15, lhalf, acc);
  }
#endif

  // Epilogue. C layout: vgpr r -> row r + 8*lhalf; col = lane&15.
#pragma unroll
  for (int sm = 0; sm < 4; sm++) {
#pragma unroll
    for (int sn = 0; sn < 4; sn++) {
      const int col = bn + waveN + sn * 16 + l15;
      const float bv = bias[col];
#pragma unroll
      for (int r = 0; r < 8; r++) {
        const int row = bm + waveM + sm * 16 + r + lhalf * 8;
        float v = acc[sm][sn][r] + bv;
        if (RES) v += res[(size_t)row * N + col];
        if (RELU) v = fmaxf(v, 0.0f);
        if (OUTBF)
          Cb[(size_t)row * N + col] = (__bf16)v;
        else
          Cf[(size_t)row * N + col] = v;
      }
    }
  }
}

// ---------------------------------------------------------------------------
// Transpose V (from fused QKV buffer, col base 2048) into VT[B,H,DH,S] bf16
// ---------------------------------------------------------------------------
__global__ __launch_bounds__(256) void transpose_v_k(
    const __bf16* __restrict__ QKV, __bf16* __restrict__ VT) {
  __shared__ __bf16 tile[32][33];
  const int bh = blockIdx.z;
  const int b = bh >> 4, h = bh & 15;
  const int s0 = blockIdx.x * 32;
  const int d0 = blockIdx.y * 32;
  const int tx = threadIdx.x & 31;
  const int ty = threadIdx.x >> 5;
#pragma unroll
  for (int i = 0; i < 32; i += 8)
    tile[ty + i][tx] =
        QKV[(size_t)(b * SEQ + s0 + ty + i) * 3072 + 2048 + h * DHEAD + d0 + tx];
  __syncthreads();
#pragma unroll
  for (int i = 0; i < 32; i += 8)
    VT[(size_t)(bh * DHEAD + d0 + ty + i) * SEQ + s0 + tx] = tile[tx][ty + i];
}

// ---------------------------------------------------------------------------
// Flash-style attention. Grid: (S/128, B*H), 256 threads = 8 waves.
// Each wave owns 16 query rows; 32 keys per step:
//   4 x WMMA (Q*K^T) -> online softmax (DPP reductions) -> P via LDS ->
//   4 x WMMA (P*V).  No pre-softmax scale; O /= (l * sqrt(DH)) at the end.
// ---------------------------------------------------------------------------
__global__ __launch_bounds__(256) void attention_k(
    const __bf16* __restrict__ QKV, const __bf16* __restrict__ VT,
    __bf16* __restrict__ Z) {
  __shared__ __bf16 sP[8 * 16 * 40];  // per-wave 16x32 P tile, rows padded to 40
  const int wave = threadIdx.x >> 5;
  const int lane = threadIdx.x & 31;
  const int l15 = lane & 15;
  const int lhalf = lane >> 4;
  const int bh = blockIdx.y;
  const int b = bh >> 4, h = bh & 15;
  const int qbase = blockIdx.x * 128 + wave * 16;
  const size_t ld = 3072;

  const __bf16* Qp = QKV + (size_t)(b * SEQ + qbase) * ld + h * DHEAD;
  const __bf16* Kp = QKV + (size_t)(b * SEQ) * ld + DMODEL + h * DHEAD;
  const __bf16* Vt = VT + (size_t)bh * DHEAD * SEQ;

  FragBF aQ[2];
#pragma unroll
  for (int c = 0; c < 2; c++) {
    const __bf16* p = Qp + (size_t)l15 * ld + c * 32 + lhalf * 8;
    aQ[c].u[0] = *(const uint4*)p;
    aQ[c].u[1] = *(const uint4*)(p + 16);
  }

  v8f O[4];
#pragma unroll
  for (int nt = 0; nt < 4; nt++) O[nt] = v8f_zero();
  float mrun[8], lrun[8];
#pragma unroll
  for (int r = 0; r < 8; r++) { mrun[r] = -3.0e38f; lrun[r] = 0.0f; }

  __bf16* myP = sP + wave * (16 * 40);

  for (int j0 = 0; j0 < SEQ; j0 += 32) {
    FragBF bK[2][2];
#pragma unroll
    for (int t = 0; t < 2; t++)
#pragma unroll
      for (int c = 0; c < 2; c++) {
        const __bf16* p =
            Kp + (size_t)(j0 + 16 * t + l15) * ld + c * 32 + lhalf * 8;
        bK[t][c].u[0] = *(const uint4*)p;
        bK[t][c].u[1] = *(const uint4*)(p + 16);
      }

    v8f s0 = wmma_bf16(aQ[0].v, bK[0][0].v, v8f_zero());
    s0 = wmma_bf16(aQ[1].v, bK[0][1].v, s0);
    v8f s1 = wmma_bf16(aQ[0].v, bK[1][0].v, v8f_zero());
    s1 = wmma_bf16(aQ[1].v, bK[1][1].v, s1);

    // online softmax bookkeeping (row r <-> vgpr r + 8*lhalf, stats per lane)
    float nm[8];
#pragma unroll
    for (int r = 0; r < 8; r++) {
      float mx = red16_max(fmaxf(s0[r], s1[r]));
      nm[r] = fmaxf(mrun[r], mx);
      float sc = __expf(mrun[r] - nm[r]);
      lrun[r] *= sc;
      mrun[r] = nm[r];
#pragma unroll
      for (int nt = 0; nt < 4; nt++) O[nt][r] *= sc;
    }
#pragma unroll
    for (int r = 0; r < 8; r++) {
      float p0 = __expf(s0[r] - nm[r]);
      float p1 = __expf(s1[r] - nm[r]);
      s0[r] = p0; s1[r] = p1;
      lrun[r] += red16_sum(p0 + p1);
    }

    // scatter P (C layout) into LDS as row-major 16x32, read back as A frag
#pragma unroll
    for (int r = 0; r < 8; r++) {
      myP[(r + lhalf * 8) * 40 + l15]      = (__bf16)s0[r];
      myP[(r + lhalf * 8) * 40 + 16 + l15] = (__bf16)s1[r];
    }
    FragBF aP;
    {
      const __bf16* p = myP + l15 * 40 + lhalf * 8;
      aP.u[0] = *(const uint4*)p;
      aP.u[1] = *(const uint4*)(p + 16);
    }

    FragBF bV[4];
#pragma unroll
    for (int nt = 0; nt < 4; nt++) {
      const __bf16* p = Vt + (size_t)(nt * 16 + l15) * SEQ + j0 + lhalf * 8;
      bV[nt].u[0] = *(const uint4*)p;
      bV[nt].u[1] = *(const uint4*)(p + 16);
    }
#pragma unroll
    for (int nt = 0; nt < 4; nt++) O[nt] = wmma_bf16(aP.v, bV[nt].v, O[nt]);

    if (j0 + 32 < SEQ) {
      __builtin_prefetch(Kp + (size_t)(j0 + 32 + l15) * ld, 0, 1);
      __builtin_prefetch(Vt + (size_t)l15 * SEQ + j0 + 32, 0, 1);
    }
  }

#pragma unroll
  for (int r = 0; r < 8; r++) {
    const float inv = 1.0f / (lrun[r] * 8.0f);  // sqrt(64) = 8
    const size_t row = (size_t)(b * SEQ + qbase + r + lhalf * 8);
#pragma unroll
    for (int nt = 0; nt < 4; nt++)
      Z[row * DMODEL + h * DHEAD + nt * 16 + l15] = (__bf16)(O[nt][r] * inv);
  }
}

// ---------------------------------------------------------------------------
// LayerNorm over D=1024; one block (256 threads) per row; 4 elems/thread.
// ---------------------------------------------------------------------------
template <bool WB>
__global__ __launch_bounds__(256) void layernorm_k(
    const float* __restrict__ X, const float* __restrict__ g,
    const float* __restrict__ be, float* __restrict__ Yf,
    __bf16* __restrict__ Yb) {
  const int row = blockIdx.x;
  const float* x = X + (size_t)row * DMODEL;
  float4 v = ((const float4*)x)[threadIdx.x];
  float s1 = v.x + v.y + v.z + v.w;
  float s2 = v.x * v.x + v.y * v.y + v.z * v.z + v.w * v.w;
#pragma unroll
  for (int m = 16; m >= 1; m >>= 1) {
    s1 += __shfl_xor(s1, m, 32);
    s2 += __shfl_xor(s2, m, 32);
  }
  __shared__ float r1[8], r2[8];
  const int wave = threadIdx.x >> 5, lane = threadIdx.x & 31;
  if (lane == 0) { r1[wave] = s1; r2[wave] = s2; }
  __syncthreads();
  float t1 = 0.f, t2 = 0.f;
#pragma unroll
  for (int i = 0; i < 8; i++) { t1 += r1[i]; t2 += r2[i]; }
  const float mu = t1 * (1.0f / DMODEL);
  const float var = t2 * (1.0f / DMODEL) - mu * mu;
  const float inv = rsqrtf(var + 1e-5f);
  float4 gv = ((const float4*)g)[threadIdx.x];
  float4 bv = ((const float4*)be)[threadIdx.x];
  float4 y;
  y.x = (v.x - mu) * inv * gv.x + bv.x;
  y.y = (v.y - mu) * inv * gv.y + bv.y;
  y.z = (v.z - mu) * inv * gv.z + bv.z;
  y.w = (v.w - mu) * inv * gv.w + bv.w;
  ((float4*)(Yf + (size_t)row * DMODEL))[threadIdx.x] = y;
  if (WB) {
    v4bf o;
    o[0] = (__bf16)y.x; o[1] = (__bf16)y.y; o[2] = (__bf16)y.z; o[3] = (__bf16)y.w;
    ((v4bf*)(Yb + (size_t)row * DMODEL))[threadIdx.x] = o;
  }
}

// ---------------------------------------------------------------------------
// Host-side orchestration
// ---------------------------------------------------------------------------
extern "C" void kernel_launch(void* const* d_in, const int* in_sizes, int n_in,
                              void* d_out, int out_size, void* d_ws,
                              size_t ws_size, hipStream_t stream) {
  const float* emb = (const float*)d_in[0];
  const float* Wq = (const float*)d_in[1];
  const float* bq = (const float*)d_in[2];
  const float* Wk = (const float*)d_in[3];
  const float* bk = (const float*)d_in[4];
  const float* Wv = (const float*)d_in[5];
  const float* bv = (const float*)d_in[6];
  const float* W0 = (const float*)d_in[7];
  const float* b0 = (const float*)d_in[8];
  const float* g1 = (const float*)d_in[9];
  const float* be1 = (const float*)d_in[10];
  const float* W1 = (const float*)d_in[11];
  const float* b1 = (const float*)d_in[12];
  const float* W2 = (const float*)d_in[13];
  const float* b2 = (const float*)d_in[14];
  const float* g2 = (const float*)d_in[15];
  const float* be2 = (const float*)d_in[16];
  float* out = (float*)d_out;

  char* ws = (char*)d_ws;
  auto alloc = [&](size_t bytes) -> char* {
    char* p = ws;
    ws += (bytes + 255) & ~(size_t)255;
    return p;
  };
  __bf16* embB = (__bf16*)alloc((size_t)MROWS * DMODEL * 2);
  __bf16* WqkvB = (__bf16*)alloc((size_t)3 * DMODEL * DMODEL * 2);
  float* bqkv = (float*)alloc((size_t)3 * DMODEL * 4);
  __bf16* W0b = (__bf16*)alloc((size_t)DMODEL * DMODEL * 2);
  __bf16* W1b = (__bf16*)alloc((size_t)DFF * DMODEL * 2);
  __bf16* W2b = (__bf16*)alloc((size_t)DMODEL * DFF * 2);
  __bf16* QKVb = (__bf16*)alloc((size_t)MROWS * 3072 * 2);
  __bf16* VTb = (__bf16*)alloc((size_t)BATCH * NHEAD * DHEAD * SEQ * 2);
  __bf16* Zb = (__bf16*)alloc((size_t)MROWS * DMODEL * 2);
  float* pre1 = (float*)alloc((size_t)MROWS * DMODEL * 4);
  float* out1f = (float*)alloc((size_t)MROWS * DMODEL * 4);
  __bf16* out1b = (__bf16*)alloc((size_t)MROWS * DMODEL * 2);
  __bf16* Hb = (__bf16*)alloc((size_t)MROWS * DFF * 2);
  float* pre2 = (float*)alloc((size_t)MROWS * DMODEL * 4);

  auto cvt = [&](const float* src, __bf16* dst, long n) {
    long n4 = n / 4;
    f32_to_bf16_k<<<(unsigned)((n4 + 255) / 256), 256, 0, stream>>>(src, dst, n4);
  };
  cvt(emb, embB, (long)MROWS * DMODEL);
  cvt(Wq, WqkvB + 0 * (size_t)DMODEL * DMODEL, (long)DMODEL * DMODEL);
  cvt(Wk, WqkvB + 1 * (size_t)DMODEL * DMODEL, (long)DMODEL * DMODEL);
  cvt(Wv, WqkvB + 2 * (size_t)DMODEL * DMODEL, (long)DMODEL * DMODEL);
  cvt(W0, W0b, (long)DMODEL * DMODEL);
  cvt(W1, W1b, (long)DFF * DMODEL);
  cvt(W2, W2b, (long)DMODEL * DFF);
  copy_f32_k<<<4, 256, 0, stream>>>(bq, bqkv + 0 * DMODEL, DMODEL);
  copy_f32_k<<<4, 256, 0, stream>>>(bk, bqkv + 1 * DMODEL, DMODEL);
  copy_f32_k<<<4, 256, 0, stream>>>(bv, bqkv + 2 * DMODEL, DMODEL);

  // 1) fused QKV projection: [4096,1024] x [3072,1024]^T -> bf16 [4096,3072]
  gemm_bf16_k<false, false, true><<<dim3(MROWS / 128, 3072 / 256), 256, 0,
                                    stream>>>(embB, WqkvB, bqkv, nullptr,
                                              nullptr, QKVb, MROWS, 3072,
                                              DMODEL);
  // 2) transpose V into [B,H,DH,S]
  transpose_v_k<<<dim3(SEQ / 32, DHEAD / 32, BATCH * NHEAD), 256, 0, stream>>>(
      QKVb, VTb);
  // 3) attention -> Z bf16 [4096,1024]
  attention_k<<<dim3(SEQ / 128, BATCH * NHEAD), 256, 0, stream>>>(QKVb, VTb, Zb);
  // 4) output projection + residual (emb) -> pre-LN1 f32
  gemm_bf16_k<false, true, false><<<dim3(MROWS / 128, DMODEL / 256), 256, 0,
                                    stream>>>(Zb, W0b, b0, emb, pre1, nullptr,
                                              MROWS, DMODEL, DMODEL);
  // 5) LayerNorm1 -> out1 (f32 + bf16)
  layernorm_k<true><<<MROWS, 256, 0, stream>>>(pre1, g1, be1, out1f, out1b);
  // 6) FFN1 with ReLU -> H bf16 [4096,4096]
  gemm_bf16_k<true, false, true><<<dim3(MROWS / 128, DFF / 256), 256, 0,
                                   stream>>>(out1b, W1b, b1, nullptr, nullptr,
                                             Hb, MROWS, DFF, DMODEL);
  // 7) FFN2 + residual (out1) -> pre-LN2 f32
  gemm_bf16_k<false, true, false><<<dim3(MROWS / 128, DMODEL / 256), 256, 0,
                                    stream>>>(Hb, W2b, b2, out1f, pre2, nullptr,
                                              MROWS, DMODEL, DFF);
  // 8) LayerNorm2 -> final output f32
  layernorm_k<false><<<MROWS, 256, 0, stream>>>(pre2, g2, be2, out, nullptr);
}
